// AttentionCostVolume_40604620816731
// MI455X (gfx1250) — matrix operand order repaired and checked by reading, not compiled, not run on hardware
//
#include <hip/hip_runtime.h>
#include <hip/hip_bf16.h>
#include <math.h>

// ---------------------------------------------------------------------------
// AttentionCostVolume on MI455X (gfx1250, wave32, WMMA 16x16x32 f16)
//
//  K0  normalize f1 over channels                     -> x1n   (fp32, 16.8MB)
//  K1  17x17 cost volume + leaky_relu(0.1)            -> mv    (fp32, 37.9MB)
//  K2  7x7 depthwise att conv + bias, gate mv         -> attv  (fp16, 19MB)
//  P   pack conv weights fp32 -> WMMA-A-fragment fp16 -> w1f/w2f
//  K3  conv 289->144 3x3 implicit GEMM (WMMA) +BN+GELU-> xh    (fp16, 9.4MB)
//  K4  conv 144->49  3x3 implicit GEMM (WMMA) +BN+GELU-> d_out (fp32)
//
// GEMM kernels use two-stage LDS staging: halo -> fragment-ordered im2col,
// so the K-loop is pure {A: 32B global load, B: 32B ds load, v_wmma}.
// ---------------------------------------------------------------------------

typedef __attribute__((ext_vector_type(16))) _Float16 v16h;
typedef __attribute__((ext_vector_type(8)))  float    v8f;

#define HDIM   128
#define WDIM   128
#define HW     16384      // 128*128
#define BS     2
#define CIN    128        // f1/f2 channels
#define SR     8
#define MOFF   17
#define NPLANE 289        // 17*17
#define MID    144
#define OUTP   49
#define K1DIM  2601       // 289*9
#define K2DIM  1296       // 144*9
#define KK1    82         // ceil(2601/32)
#define KK2    41         // ceil(1296/32)
#define MT1    9          // 144/16
#define MT2    4          // ceil(49/16)

// ---------------------------------------------------------------------------
// K0: x1n[b,c,p] = f1[b,c,p] / max(||f1[b,:,p]||_2, 1e-12)
// ---------------------------------------------------------------------------
__global__ __launch_bounds__(256) void k_normalize(const float* __restrict__ f1,
                                                   float* __restrict__ x1n) {
  int idx = blockIdx.x * 256 + threadIdx.x;          // 0 .. BS*HW-1
  if (idx >= BS * HW) return;
  int b = idx >> 14, p = idx & (HW - 1);
  const float* src = f1 + ((size_t)b * CIN) * HW + p;
  float ss = 0.f;
  #pragma unroll 4
  for (int c = 0; c < CIN; ++c) { float v = src[c * HW]; ss += v * v; }
  float inv = 1.0f / fmaxf(sqrtf(ss), 1e-12f);
  float* dst = x1n + ((size_t)b * CIN) * HW + p;
  #pragma unroll 4
  for (int c = 0; c < CIN; ++c) dst[c * HW] = src[c * HW] * inv;
}

// ---------------------------------------------------------------------------
// K1: mv[b,off,p] = leaky( mean_c( x1n[b,c,p] * f2pad[b,c,p+off] ), 0.1 )
// ---------------------------------------------------------------------------
__global__ __launch_bounds__(256) void k_costvol(const float* __restrict__ x1n,
                                                 const float* __restrict__ f2,
                                                 float* __restrict__ mv) {
  int idx = blockIdx.x * 256 + threadIdx.x;          // 0 .. BS*NPLANE*HW-1
  if (idx >= BS * NPLANE * HW) return;
  int p = idx & (HW - 1);
  int t = idx >> 14;
  int off = t % NPLANE;
  int b   = t / NPLANE;
  int h = p >> 7, w = p & 127;
  int dy = off / MOFF - SR, dx = off % MOFF - SR;
  int h2 = h + dy, w2 = w + dx;
  float cv = 0.f;
  if (h2 >= 0 && h2 < HDIM && w2 >= 0 && w2 < WDIM) {
    const float* a = x1n + ((size_t)b * CIN) * HW + p;
    const float* c = f2  + ((size_t)b * CIN) * HW + (h2 << 7) + w2;
    float s = 0.f;
    #pragma unroll 4
    for (int ci = 0; ci < CIN; ++ci) s += a[ci * HW] * c[ci * HW];
    cv = s * (1.0f / CIN);
  }
  mv[idx] = cv > 0.f ? cv : 0.1f * cv;
}

// ---------------------------------------------------------------------------
// K2: att = depthwise7x7(mv) + bias ; attv = fp16(mv * att)
// ---------------------------------------------------------------------------
__global__ __launch_bounds__(256) void k_attention(const float* __restrict__ mv,
                                                   const float* __restrict__ aw,
                                                   const float* __restrict__ ab,
                                                   _Float16* __restrict__ attv) {
  int idx = blockIdx.x * 256 + threadIdx.x;
  if (idx >= BS * NPLANE * HW) return;
  int p = idx & (HW - 1);
  int t = idx >> 14;
  int ci = t % NPLANE;
  int h = p >> 7, w = p & 127;
  const float* plane = mv + ((size_t)(idx - p));     // start of [b,ci] plane
  const float* wk = aw + ci * 49;
  float att = ab[ci];
  #pragma unroll
  for (int ty = 0; ty < 7; ++ty) {
    int hh = h + ty - 3;
    if (hh < 0 || hh >= HDIM) continue;
    #pragma unroll
    for (int tx = 0; tx < 7; ++tx) {
      int ww = w + tx - 3;
      if (ww < 0 || ww >= WDIM) continue;
      att += wk[ty * 7 + tx] * plane[(hh << 7) + ww];
    }
  }
  attv[idx] = (_Float16)(plane[p] * att);
}

// ---------------------------------------------------------------------------
// Weight pre-pack: fp32 [CO][K] -> fp16 WMMA A fragments in contiguous k order:
//   el 0..15 -> K = kk*32 + (lane>>4)*16 + el
// Stored at wf[((mt*KK + kk)*32 + lane)*16 + el]  (contiguous 32B per lane).
// ---------------------------------------------------------------------------
__global__ __launch_bounds__(256) void k_pack(const float* __restrict__ w,
                                              _Float16* __restrict__ wf,
                                              int CO, int K, int KK, int total) {
  int i = blockIdx.x * 256 + threadIdx.x;
  if (i >= total) return;
  int el   = i & 15;
  int lane = (i >> 4) & 31;
  int rest = i >> 9;
  int kk   = rest % KK;
  int mt   = rest / KK;
  int row  = mt * 16 + (lane & 15);
  int hi   = lane >> 4;
  int k    = kk * 32 + hi * 16 + el;
  float v  = (row < CO && k < K) ? w[(size_t)row * K + k] : 0.f;
  wf[i] = (_Float16)v;
}

// ---------------------------------------------------------------------------
// Implicit-GEMM 3x3 conv via WMMA, fused BN + exact GELU epilogue.
// Block = NW waves; one 16-wide pixel strip (single row; 128%16==0).
// Stage 1: halo [Cin][3][18] global -> LDS (zero-padded).
// Stage 2: expand halo -> fragment-ordered B in LDS: sBf[kk][lane][16] where
//          lane = (kInTile>=16)*16 + col, el = k&15 (contiguous k order, same as A:
//          lanes 0-15 hold K=kk*32..+15, lanes 16-31 hold K=kk*32+16..+31).
// Main loop per wave: contiguous 32B A (global) + 32B B (LDS) + v_wmma.
// ---------------------------------------------------------------------------
template <int CINCH, int KDIM, int KK, int NW, bool OUT_F32>
__global__ void k_conv3x3_wmma(const _Float16* __restrict__ src,   // [BS][CINCH][HW] fp16
                               const _Float16* __restrict__ wf,    // packed fragments
                               const float* __restrict__ bg, const float* __restrict__ bb,
                               const float* __restrict__ bm, const float* __restrict__ bv,
                               _Float16* __restrict__ dst_h,       // fp16 out (or null)
                               float* __restrict__ dst_f,          // fp32 out (or null)
                               int CO) {
  __shared__ _Float16 sHalo[CINCH * 3 * 18];
  __shared__ v16h     sBf[KK * 32];                 // fragment-ordered im2col
  const int n0 = blockIdx.x * 16;                   // flat (b,h,w0)
  const int b  = n0 >> 14;
  const int h  = (n0 >> 7) & 127;
  const int w0 = n0 & 127;

  // ---- stage 1: halo (rows h-1..h+1, cols w0-1..w0+16, all Cin) ----
  for (int i = threadIdx.x; i < CINCH * 3 * 18; i += NW * 32) {
    int ww = i % 18;
    int t  = i / 18;
    int r  = t % 3;
    int ci = t / 3;
    int hh = h + r - 1, wc = w0 + ww - 1;
    _Float16 v = (_Float16)0.f;
    if (hh >= 0 && hh < HDIM && wc >= 0 && wc < WDIM)
      v = src[(((size_t)b * CINCH + ci) << 14) + (hh << 7) + wc];
    sHalo[i] = v;
  }
  __syncthreads();

  // ---- stage 2: expand to fragment order (tail k>=KDIM zero-filled) ----
  {
    _Float16* sBfh = (_Float16*)sBf;
    for (int i = threadIdx.x; i < KK * 32 * 16; i += NW * 32) {
      unsigned k   = (unsigned)i >> 4;              // 0 .. KK*32-1
      unsigned col = (unsigned)i & 15;
      unsigned kk  = k >> 5;
      unsigned hi  = (k >> 4) & 1;
      unsigned el  = k & 15;
      _Float16 v = (_Float16)0.f;
      if (k < (unsigned)KDIM) {
        unsigned ci  = k / 9u;                      // const-divisor mul/shift
        unsigned tap = k - ci * 9u;
        unsigned ty  = tap / 3u;
        unsigned tx  = tap - ty * 3u;
        v = sHalo[(ci * 3u + ty) * 18u + col + tx];
      }
      sBfh[(kk * 32u + hi * 16u + col) * 16u + el] = v;
    }
  }
  __syncthreads();

  // ---- main GEMM loop: one 16x16 tile per wave ----
  const int wv   = threadIdx.x >> 5;
  const int lane = threadIdx.x & 31;

  v8f acc = {};
  const v16h* afr = (const v16h*)wf + ((size_t)wv * KK) * 32 + lane;
  const v16h* bfr = sBf + lane;
  for (int kk = 0; kk < KK; ++kk) {
    v16h a = afr[kk * 32];                          // contiguous 32B/lane (global)
    v16h bf = bfr[kk * 32];                         // contiguous 32B/lane (LDS)
    __builtin_prefetch(&afr[(kk + 1) * 32], 0, 0);  // speculative, safe OOB
    acc = __builtin_amdgcn_wmma_f32_16x16x32_f16(false, a, false, bf,
                                                 (short)0, acc, false, false);
  }

  // ---- epilogue: BN (eval) + exact GELU ----
  const int col = lane & 15;
  const int hi  = lane >> 4;
  const int wcol = w0 + col;
  #pragma unroll
  for (int r = 0; r < 8; ++r) {
    int co = wv * 16 + hi * 8 + r;                  // D layout: VGPR r -> M
    if (co >= CO) continue;
    float sc = bg[co] * rsqrtf(bv[co] + 1e-5f);
    float y  = (acc[r] - bm[co]) * sc + bb[co];
    float g  = 0.5f * y * (1.0f + erff(y * 0.70710678118654752f));
    size_t o = (((size_t)b * CO + co) << 14) + (h << 7) + wcol;
    if (OUT_F32) dst_f[o] = g; else dst_h[o] = (_Float16)g;
  }
}

// ---------------------------------------------------------------------------
extern "C" void kernel_launch(void* const* d_in, const int* in_sizes, int n_in,
                              void* d_out, int out_size, void* d_ws, size_t ws_size,
                              hipStream_t stream) {
  const float* f1   = (const float*)d_in[0];
  const float* f2   = (const float*)d_in[1];
  const float* aw   = (const float*)d_in[2];
  const float* ab   = (const float*)d_in[3];
  const float* c1w  = (const float*)d_in[4];
  const float* bn1g = (const float*)d_in[5];
  const float* bn1b = (const float*)d_in[6];
  const float* bn1m = (const float*)d_in[7];
  const float* bn1v = (const float*)d_in[8];
  const float* c2w  = (const float*)d_in[9];
  const float* bn2g = (const float*)d_in[10];
  const float* bn2b = (const float*)d_in[11];
  const float* bn2m = (const float*)d_in[12];
  const float* bn2v = (const float*)d_in[13];
  float* out = (float*)d_out;

  // carve workspace (256B aligned slices)
  char* ws = (char*)d_ws;
  size_t off = 0;
  auto take = [&](size_t bytes) { char* p = ws + off; off = (off + bytes + 255) & ~(size_t)255; return p; };
  float*    x1n  = (float*)take((size_t)BS * CIN * HW * 4);
  float*    mv   = (float*)take((size_t)BS * NPLANE * HW * 4);
  _Float16* attv = (_Float16*)take((size_t)BS * NPLANE * HW * 2);
  _Float16* xh   = (_Float16*)take((size_t)BS * MID * HW * 2);
  _Float16* w1f  = (_Float16*)take((size_t)MT1 * KK1 * 32 * 16 * 2);
  _Float16* w2f  = (_Float16*)take((size_t)MT2 * KK2 * 32 * 16 * 2);
  (void)ws_size;

  // weight pre-pack (independent; run first)
  {
    int tot1 = MT1 * KK1 * 32 * 16;
    k_pack<<<(tot1 + 255) / 256, 256, 0, stream>>>(c1w, w1f, MID, K1DIM, KK1, tot1);
    int tot2 = MT2 * KK2 * 32 * 16;
    k_pack<<<(tot2 + 255) / 256, 256, 0, stream>>>(c2w, w2f, OUTP, K2DIM, KK2, tot2);
  }

  k_normalize<<<(BS * HW + 255) / 256, 256, 0, stream>>>(f1, x1n);

  {
    int tot = BS * NPLANE * HW;
    k_costvol<<<(tot + 255) / 256, 256, 0, stream>>>(x1n, f2, mv);
    k_attention<<<(tot + 255) / 256, 256, 0, stream>>>(mv, aw, ab, attv);
  }

  // conv1: 289->144, 9 M-tiles => 9 waves/block, fp16 out for conv2
  k_conv3x3_wmma<NPLANE, K1DIM, KK1, MT1, false>
      <<<BS * HW / 16, MT1 * 32, 0, stream>>>(attv, w1f, bn1g, bn1b, bn1m, bn1v,
                                              xh, nullptr, MID);

  // conv2: 144->49, 4 M-tiles => 4 waves/block, fp32 out to d_out
  k_conv3x3_wmma<MID, K2DIM, KK2, MT2, true>
      <<<BS * HW / 16, MT2 * 32, 0, stream>>>(xh, w2f, bn2g, bn2b, bn2m, bn2v,
                                              nullptr, out, OUTP);
}
